// GNNSelectionNetwork_35699768165082
// MI455X (gfx1250) — matrix operand order repaired and checked by reading, not compiled, not run on hardware
//
#include <hip/hip_runtime.h>
#include <math.h>

#define BB 32
#define TT 10
#define NN 64
#define DD 4
#define HH 64

typedef __attribute__((ext_vector_type(16))) _Float16 v16h;
typedef __attribute__((ext_vector_type(8)))  float    v8f;
typedef __attribute__((ext_vector_type(2)))  _Float16 h2;

// ---------------------------------------------------------------------------
// WMMA helpers (wave32, 16x16x32 f16 -> f32)
// ---------------------------------------------------------------------------
__device__ __forceinline__ v8f wmma16x16x32(v16h a, v16h b, v8f c) {
  // 8 args: (neg_a, A, neg_b, B, c_mod, C, reuse_a, reuse_b)
  return __builtin_amdgcn_wmma_f32_16x16x32_f16(false, a, false, b, (short)0, c,
                                                false, false);
}

// A fragment (16x32 f16) gathered from row-major LDS buffer per ISA layout:
// lanes 0-15: M=lane, v0..3 -> K=0..7, v4..7 -> K=16..23
// lanes 16-31: M=lane-16, v0..3 -> K=8..15, v4..7 -> K=24..31
__device__ __forceinline__ v16h load_A_frag(const _Float16* p0, int k0, int stride) {
  int lane = threadIdx.x & 31;
  int m    = lane & 15;
  int half = lane >> 4;
  const _Float16* p = p0 + m * stride + k0;
  v16h a;
#pragma unroll
  for (int v = 0; v < 8; ++v) {
    int k = (v >> 2) * 16 + half * 8 + (v & 3) * 2;
    a[2 * v]     = p[k];
    a[2 * v + 1] = p[k + 1];
  }
  return a;
}

// B fragment: pre-packed so each lane reads 16 contiguous halves (32B, coalesced)
__device__ __forceinline__ v16h load_B_frag(const _Float16* g, int tile) {
  int lane = threadIdx.x & 31;
  return *(const v16h*)(g + (size_t)tile * 512 + lane * 16);
}

// ---------------------------------------------------------------------------
// Weight packer: W (KxN f32 row-major) -> f16 B-fragments, tile = nt*KT + kt
// lane layout: n = nt*16 + lane%16 ; k = kt*32 + (j/8)*16 + (lane/16)*8 + j%8 , j=0..15
// ---------------------------------------------------------------------------
__global__ void pack_B(const float* __restrict__ W, _Float16* __restrict__ out,
                       int K, int N) {
  int idx = blockIdx.x * blockDim.x + threadIdx.x;
  if (idx >= K * N) return;
  int KT   = K >> 5;
  int j    = idx & 15;
  int lane = (idx >> 4) & 31;
  int tile = idx >> 9;
  int kt = tile % KT;
  int nt = tile / KT;
  int n = nt * 16 + (lane & 15);
  int k = kt * 32 + (j >> 3) * 16 + (lane >> 4) * 8 + (j & 7);
  out[idx] = (_Float16)W[k * N + n];
}

// ---------------------------------------------------------------------------
// Node GRU: 2048 rows, T=10, D=4, H=64.  256 threads = 4 rows x 64 hidden.
// ---------------------------------------------------------------------------
__global__ void gru_node(const float* __restrict__ x,
                         const float* __restrict__ Wi,
                         const float* __restrict__ Wh,
                         const float* __restrict__ bi,
                         const float* __restrict__ bh,
                         float* __restrict__ hout) {
  __shared__ float sWh[64 * 192];
  __shared__ float sWi[4 * 192];
  __shared__ float sb[2 * 192];
  __shared__ float sh[4][64];
  int tid = threadIdx.x;
  for (int i = tid; i < 64 * 192; i += 256) sWh[i] = Wh[i];
  for (int i = tid; i < 4 * 192; i += 256) sWi[i] = Wi[i];
  for (int i = tid; i < 192; i += 256) { sb[i] = bi[i]; sb[192 + i] = bh[i]; }
  int rl  = tid >> 6;
  int j   = tid & 63;
  int row = blockIdx.x * 4 + rl;      // row = b*N + n
  int b = row >> 6, n = row & 63;
  sh[rl][j] = 0.f;
  __syncthreads();

  float dscale = 0.134217728f;        // 0.8^9 ; multiplied by 1.25 each step
  float hval = 0.f;
  for (int t = 0; t < TT; ++t) {
    const float* xp = x + ((size_t)(b * TT + t) * NN + n) * DD;
    float x0 = xp[0] * dscale, x1 = xp[1] * dscale;
    float x2 = xp[2] * dscale, x3 = xp[3] * dscale;
    float ir = sb[j]       + x0 * sWi[j]       + x1 * sWi[192 + j] + x2 * sWi[384 + j] + x3 * sWi[576 + j];
    float iz = sb[64 + j]  + x0 * sWi[64 + j]  + x1 * sWi[256 + j] + x2 * sWi[448 + j] + x3 * sWi[640 + j];
    float in_= sb[128 + j] + x0 * sWi[128 + j] + x1 * sWi[320 + j] + x2 * sWi[512 + j] + x3 * sWi[704 + j];
    float hr = sb[192 + j], hz = sb[256 + j], hn = sb[320 + j];
#pragma unroll 8
    for (int k = 0; k < 64; ++k) {
      float hk = sh[rl][k];
      hr += hk * sWh[k * 192 + j];
      hz += hk * sWh[k * 192 + 64 + j];
      hn += hk * sWh[k * 192 + 128 + j];
    }
    float r  = 1.f / (1.f + expf(-(ir + hr)));
    float z  = 1.f / (1.f + expf(-(iz + hz)));
    float nn = tanhf(in_ + r * hn);
    hval = (1.f - z) * nn + z * hval;
    __syncthreads();
    sh[rl][j] = hval;
    __syncthreads();
    dscale *= 1.25f;
  }
  hout[(size_t)row * HH + j] = hval;
}

// ---------------------------------------------------------------------------
// indeg[b,r] = sum_s adj[b,s,r]  (1e-5 if zero)
// ---------------------------------------------------------------------------
__global__ void indeg_kernel(const int* __restrict__ adj, float* __restrict__ indeg) {
  int idx = blockIdx.x * blockDim.x + threadIdx.x;
  if (idx >= BB * NN) return;
  int b = idx >> 6, r = idx & 63;
  float cnt = 0.f;
  for (int s = 0; s < NN; ++s) cnt += (float)adj[(size_t)(b * NN + s) * NN + r];
  indeg[idx] = (cnt != 0.f) ? cnt : 1e-5f;
}

__global__ void zero_kernel(float* __restrict__ p, int n) {
  int idx = blockIdx.x * blockDim.x + threadIdx.x;
  if (idx < n) p[idx] = 0.f;
}

__global__ void update_kernel(float* __restrict__ h, const float* __restrict__ acc,
                              const float* __restrict__ indeg, int n) {
  int idx = blockIdx.x * blockDim.x + threadIdx.x;
  if (idx < n) h[idx] += acc[idx] / indeg[idx >> 6];
}

// ---------------------------------------------------------------------------
// Per-round precompute: P = h @ W0[0:64,:] ; Q = h @ W0[64:128,:] + b0 (f16)
// one wave / 16 rows; WMMA over K=64; C tiles bounced through LDS so the
// global P/Q writes are coalesced b128 stores (no per-half store+wait chains)
// ---------------------------------------------------------------------------
__global__ void pq_kernel(const float* __restrict__ h,
                          const _Float16* __restrict__ W0p,
                          const float* __restrict__ b0,
                          _Float16* __restrict__ P,
                          _Float16* __restrict__ Q) {
  __shared__ _Float16 shh[16 * 64];
  __shared__ _Float16 sPo[16 * 128];
  __shared__ _Float16 sQo[16 * 128];
  int lane = threadIdx.x;
  int row0 = blockIdx.x * 16;
  for (int i = lane; i < 16 * 64; i += 32)
    shh[i] = (_Float16)h[(size_t)(row0 + (i >> 6)) * HH + (i & 63)];
  __syncthreads();
  int nlo = lane & 15;
  int mhi = (lane >> 4) * 8;
#pragma unroll
  for (int nt = 0; nt < 8; ++nt) {
    v8f cP = {};
    float bq = b0[nt * 16 + nlo];
    v8f cQ;
#pragma unroll
    for (int v = 0; v < 8; ++v) cQ[v] = bq;
#pragma unroll
    for (int kt = 0; kt < 2; ++kt) {
      v16h a  = load_A_frag(shh, kt * 32, 64);
      v16h wp = load_B_frag(W0p, nt * 4 + kt);      // W0 rows 0..63
      v16h wq = load_B_frag(W0p, nt * 4 + kt + 2);  // W0 rows 64..127
      cP = wmma16x16x32(a, wp, cP);
      cQ = wmma16x16x32(a, wq, cQ);
    }
#pragma unroll
    for (int v = 0; v < 8; ++v) {
      int o = (mhi + v) * 128 + nt * 16 + nlo;
      sPo[o] = (_Float16)cP[v];
      sQo[o] = (_Float16)cQ[v];
    }
  }
  __syncthreads();
  // coalesced 16B copies LDS -> global
  const uint4* sp4 = (const uint4*)sPo;
  const uint4* sq4 = (const uint4*)sQo;
  uint4* Pg4 = (uint4*)(P + (size_t)row0 * 128);
  uint4* Qg4 = (uint4*)(Q + (size_t)row0 * 128);
  for (int i = lane; i < 256; i += 32) {
    Pg4[i] = sp4[i];
    Qg4[i] = sq4[i];
  }
}

// ---------------------------------------------------------------------------
// Message passing: one wave per (b, receiver-tile-of-16, sender-chunk-of-16).
// act1 = relu(P[s] + Q'[r]) via packed v_pk_* f16 ops; layers 2/3 via WMMA;
// mask-by-multiply register accumulation; one atomicAdd burst at the end.
// ---------------------------------------------------------------------------
__global__ void msg_round(const _Float16* __restrict__ P,
                          const _Float16* __restrict__ Q,
                          const _Float16* __restrict__ W1p,
                          const float* __restrict__ b1,
                          const _Float16* __restrict__ Wop,
                          const float* __restrict__ bo,
                          const int* __restrict__ adj,
                          float* __restrict__ acc) {
  __shared__ _Float16 sQ[16 * 128];
  __shared__ _Float16 sP[128];
  __shared__ _Float16 act1[16 * 128];
  __shared__ _Float16 act2[16 * 128];
  __shared__ float smask[16];

  int lane = threadIdx.x;
  int blk = blockIdx.x;
  int sc = blk & 3;
  int rt = (blk >> 2) & 3;
  int b  = blk >> 4;
  int r0 = rt * 16;
  int s0 = sc * 16;

  const h2* Qg = (const h2*)Q;
  const h2* Pg = (const h2*)P;
  h2* sQ2 = (h2*)sQ;
  h2* sP2 = (h2*)sP;
  h2* act1_2 = (h2*)act1;

  // stage receiver Q rows (packed dword copies)
  for (int i = lane; i < 16 * 64; i += 32)
    sQ2[i] = Qg[(size_t)(b * NN + r0 + (i >> 6)) * 64 + (i & 63)];

  float racc[4][8];
#pragma unroll
  for (int nt = 0; nt < 4; ++nt)
#pragma unroll
    for (int v = 0; v < 8; ++v) racc[nt][v] = 0.f;

  int nlo = lane & 15;
  int mhi = (lane >> 4) * 8;
  const h2 zz = {};

  for (int si = 0; si < 16; ++si) {
    int s = s0 + si;
    for (int c = lane; c < 64; c += 32)
      sP2[c] = Pg[(size_t)(b * NN + s) * 64 + c];
    if (lane < 16)
      smask[lane] = (float)adj[(size_t)(b * NN + s) * NN + r0 + lane];
    __syncthreads();

    // layer 1 (linear part precomputed): relu(P[s] + Q[r] + b0), packed f16
    for (int i = lane; i < 16 * 64; i += 32) {
      h2 v = sP2[i & 63] + sQ2[i];
      act1_2[i] = __builtin_elementwise_max(v, zz);
    }
    __syncthreads();

    // layer 2: relu(act1 @ W1 + b1)  (16x128x128)
#pragma unroll
    for (int nt = 0; nt < 8; ++nt) {
      float bb = b1[nt * 16 + nlo];
      v8f c;
#pragma unroll
      for (int v = 0; v < 8; ++v) c[v] = bb;
#pragma unroll
      for (int kt = 0; kt < 4; ++kt) {
        v16h a = load_A_frag(act1, kt * 32, 128);
        v16h w = load_B_frag(W1p, nt * 4 + kt);
        c = wmma16x16x32(a, w, c);
      }
#pragma unroll
      for (int v = 0; v < 8; ++v) {
        float xx = c[v] > 0.f ? c[v] : 0.f;
        act2[(mhi + v) * 128 + nt * 16 + nlo] = (_Float16)xx;
      }
    }
    __syncthreads();

    // layer 3: act2 @ Wo + bo  (16x128x64), mask-by-multiply accumulate
#pragma unroll
    for (int nt = 0; nt < 4; ++nt) {
      float bb = bo[nt * 16 + nlo];
      v8f c;
#pragma unroll
      for (int v = 0; v < 8; ++v) c[v] = bb;
#pragma unroll
      for (int kt = 0; kt < 4; ++kt) {
        v16h a = load_A_frag(act2, kt * 32, 128);
        v16h w = load_B_frag(Wop, nt * 4 + kt);
        c = wmma16x16x32(a, w, c);
      }
#pragma unroll
      for (int v = 0; v < 8; ++v)
        racc[nt][v] += smask[mhi + v] * c[v];   // adj in {0,1}
    }
    __syncthreads();
  }

#pragma unroll
  for (int nt = 0; nt < 4; ++nt)
#pragma unroll
    for (int v = 0; v < 8; ++v)
      atomicAdd(&acc[(size_t)(b * NN + r0 + mhi + v) * HH + nt * 16 + nlo],
                racc[nt][v]);
}

// ---------------------------------------------------------------------------
// Influence head: z=[h_i|h_ego] -> relu(128) -> relu(64) -> sigmoid(1), masked
// ---------------------------------------------------------------------------
__global__ void influence_kernel(const float* __restrict__ h,
                                 const _Float16* __restrict__ iW0p,
                                 const float* __restrict__ ib0,
                                 const _Float16* __restrict__ iW1p,
                                 const float* __restrict__ ib1,
                                 const float* __restrict__ Wo,
                                 const float* __restrict__ bo,
                                 const int* __restrict__ adj,
                                 float* __restrict__ out) {
  __shared__ _Float16 sz[16 * 128];
  __shared__ _Float16 sa1[16 * 128];
  __shared__ float sa2[16 * 64];
  int lane = threadIdx.x;
  int b  = blockIdx.x >> 2;
  int i0 = (blockIdx.x & 3) * 16;
  for (int i = lane; i < 16 * 128; i += 32) {
    int m = i >> 7, c = i & 127;
    float v = (c < 64) ? h[(size_t)(b * NN + i0 + m) * HH + c]
                       : h[(size_t)(b * NN) * HH + (c - 64)];   // ego = node 0
    sz[i] = (_Float16)v;
  }
  __syncthreads();
  int nlo = lane & 15, mhi = (lane >> 4) * 8;
#pragma unroll
  for (int nt = 0; nt < 8; ++nt) {
    float bb = ib0[nt * 16 + nlo];
    v8f c;
#pragma unroll
    for (int v = 0; v < 8; ++v) c[v] = bb;
#pragma unroll
    for (int kt = 0; kt < 4; ++kt) {
      v16h a = load_A_frag(sz, kt * 32, 128);
      v16h w = load_B_frag(iW0p, nt * 4 + kt);
      c = wmma16x16x32(a, w, c);
    }
#pragma unroll
    for (int v = 0; v < 8; ++v) {
      float xx = c[v] > 0.f ? c[v] : 0.f;
      sa1[(mhi + v) * 128 + nt * 16 + nlo] = (_Float16)xx;
    }
  }
  __syncthreads();
#pragma unroll
  for (int nt = 0; nt < 4; ++nt) {
    float bb = ib1[nt * 16 + nlo];
    v8f c;
#pragma unroll
    for (int v = 0; v < 8; ++v) c[v] = bb;
#pragma unroll
    for (int kt = 0; kt < 4; ++kt) {
      v16h a = load_A_frag(sa1, kt * 32, 128);
      v16h w = load_B_frag(iW1p, nt * 4 + kt);
      c = wmma16x16x32(a, w, c);
    }
#pragma unroll
    for (int v = 0; v < 8; ++v) {
      float xx = c[v] > 0.f ? c[v] : 0.f;
      sa2[(mhi + v) * 64 + nt * 16 + nlo] = xx;
    }
  }
  __syncthreads();
  if (lane < 16) {
    int i = i0 + lane;
    float a = bo[0];
    for (int k = 0; k < 64; ++k) a += sa2[lane * 64 + k] * Wo[k];
    float sg = 1.f / (1.f + expf(-a));
    out[b * NN + i] = (adj[(size_t)(b * NN + i) * NN] == 1) ? sg : 0.f;
  }
}

// ---------------------------------------------------------------------------
// Host-side orchestration (all on `stream`, graph-capture safe)
// ---------------------------------------------------------------------------
extern "C" void kernel_launch(void* const* d_in, const int* in_sizes, int n_in,
                              void* d_out, int out_size, void* d_ws, size_t ws_size,
                              hipStream_t stream) {
  (void)in_sizes; (void)n_in; (void)out_size; (void)ws_size;
  const float* x      = (const float*)d_in[0];
  const int*   adj    = (const int*)d_in[1];
  const float* gn_Wi  = (const float*)d_in[2];
  const float* gn_Wh  = (const float*)d_in[3];
  const float* gn_bi  = (const float*)d_in[4];
  const float* gn_bh  = (const float*)d_in[5];
  // d_in[6..13]: gd_*, gc_* -> dead code in the reference (edge_enc is deleted)
  const float* msg_W0 = (const float*)d_in[14];
  const float* msg_b0 = (const float*)d_in[15];
  const float* msg_W1 = (const float*)d_in[16];
  const float* msg_b1 = (const float*)d_in[17];
  const float* msg_Wo = (const float*)d_in[18];
  const float* msg_bo = (const float*)d_in[19];
  const float* inf_W0 = (const float*)d_in[20];
  const float* inf_b0 = (const float*)d_in[21];
  const float* inf_W1 = (const float*)d_in[22];
  const float* inf_b1 = (const float*)d_in[23];
  const float* inf_Wo = (const float*)d_in[24];
  const float* inf_bo = (const float*)d_in[25];

  char* ws = (char*)d_ws;
  float*    h     = (float*)(ws + 0);          // 2048*64*4  = 512KB
  float*    acc   = (float*)(ws + 524288);     // 512KB
  float*    indeg = (float*)(ws + 1048576);    // 8KB
  _Float16* P     = (_Float16*)(ws + 1056768); // 2048*128*2 = 512KB
  _Float16* Q     = (_Float16*)(ws + 1581056); // 512KB
  _Float16* W0p   = (_Float16*)(ws + 2105344); // 32KB
  _Float16* W1p   = (_Float16*)(ws + 2138112); // 32KB
  _Float16* Wop   = (_Float16*)(ws + 2170880); // 16KB
  _Float16* iW0p  = (_Float16*)(ws + 2187264); // 32KB
  _Float16* iW1p  = (_Float16*)(ws + 2220032); // 16KB

  pack_B<<<(128 * 128 + 255) / 256, 256, 0, stream>>>(msg_W0, W0p, 128, 128);
  pack_B<<<(128 * 128 + 255) / 256, 256, 0, stream>>>(msg_W1, W1p, 128, 128);
  pack_B<<<(128 * 64 + 255) / 256, 256, 0, stream>>>(msg_Wo, Wop, 128, 64);
  pack_B<<<(128 * 128 + 255) / 256, 256, 0, stream>>>(inf_W0, iW0p, 128, 128);
  pack_B<<<(128 * 64 + 255) / 256, 256, 0, stream>>>(inf_W1, iW1p, 128, 64);

  gru_node<<<512, 256, 0, stream>>>(x, gn_Wi, gn_Wh, gn_bi, gn_bh, h);
  indeg_kernel<<<8, 256, 0, stream>>>(adj, indeg);

  for (int r = 0; r < 3; ++r) {
    pq_kernel<<<128, 32, 0, stream>>>(h, W0p, msg_b0, P, Q);
    zero_kernel<<<512, 256, 0, stream>>>(acc, BB * NN * HH);
    msg_round<<<512, 32, 0, stream>>>(P, Q, W1p, msg_b1, Wop, msg_bo, adj, acc);
    update_kernel<<<512, 256, 0, stream>>>(h, acc, indeg, BB * NN * HH);
  }

  influence_kernel<<<128, 32, 0, stream>>>(h, iW0p, inf_b0, iW1p, inf_b1,
                                           inf_Wo, inf_bo, adj, (float*)d_out);
}
